// MultiHeadAttention_67405216744307
// MI455X (gfx1250) — matrix operand order repaired and checked by reading, not compiled, not run on hardware
//
#include <hip/hip_runtime.h>

// ---------------------------------------------------------------------------
// MultiHeadAttention forward for MI455X (gfx1250, wave32, WMMA bf16 path)
// B=2, S=2048, d_model=1024, H=16, d_k=64. fp32 in/out, bf16 matmuls with
// fp32 accumulation via v_wmma_f32_16x16x32_bf16. Weight tiles are staged
// into LDS with the Tensor Data Mover (tensor_load_to_lds + s_wait_tensorcnt),
// double buffered.
// ---------------------------------------------------------------------------

#define D_MODEL 1024
#define N_HEADS 16
#define D_K     64
#define BATCH   2
#define SEQ     2048
#define TOKENS  (BATCH * SEQ)

typedef __attribute__((ext_vector_type(16))) __bf16 v16bf;
typedef __attribute__((ext_vector_type(8)))  __bf16 v8bf;
typedef __attribute__((ext_vector_type(4)))  __bf16 v4bf;
typedef __attribute__((ext_vector_type(8)))  float  v8f;
typedef __attribute__((ext_vector_type(4)))  unsigned int v4u;
typedef __attribute__((ext_vector_type(8)))  int v8i;
typedef __attribute__((ext_vector_type(4)))  int v4i;

union FragBf {
    v16bf v;
    v8bf  h[2];
};

// Build a 16-element bf16 A/B fragment from two 16-byte runs.
// A-fragment (16x32, ISA 7.12.2): lane L (m = L&15, hf = L>>4) holds
//   K = hf*8 .. hf*8+7  and  K = 16+hf*8 .. 16+hf*8+7   -> runs at +0 and +16
// B-fragment (32x16): lane L (n = L&15, hf = L>>4) holds same K as A-fragment
//   -> runs at +0 and +16
__device__ __forceinline__ v16bf load_frag(const __bf16* p0, const __bf16* p1) {
    FragBf f;
    f.h[0] = *(const v8bf*)p0;
    f.h[1] = *(const v8bf*)p1;
    return f.v;
}

// ---------------------------------------------------------------------------
// Tensor Data Mover: DMA a [rows x 32] bf16 tile (row stride = 1024 elements)
// from global into LDS, packed contiguously as [row][32]. D# built per
// CDNA5 ISA section 8 (group0: count/lds/global/type, group1: dims+strides).
// ---------------------------------------------------------------------------
#if defined(__has_builtin)
#if __has_builtin(__builtin_amdgcn_tensor_load_to_lds)
#define USE_TDM 1
#endif
#endif

#ifdef USE_TDM
__device__ __forceinline__ void tdm_load_tile(unsigned lds_addr,
                                              unsigned long long gaddr,
                                              unsigned rows) {
    const unsigned td0   = D_MODEL;      // tensor_dim0 (elements per row)
    const unsigned td1   = D_MODEL;      // tensor_dim1 (rows)
    const unsigned tile0 = 32;           // tile_dim0 (k elements)
    const unsigned tile1 = rows;         // tile_dim1 (n rows)
    const unsigned long long s0 = D_MODEL;  // tensor_dim0_stride (elements)

    v4u g0;
    g0[0] = 1u;                                               // count=1, user mode
    g0[1] = lds_addr;                                         // lds_addr [63:32]
    g0[2] = (unsigned)gaddr;                                  // global_addr lo
    g0[3] = ((unsigned)(gaddr >> 32) & 0x1FFFFFFu) | (2u << 30); // addr hi | type=2

    v8i g1;
    g1[0] = (int)(1u << 16);                                  // data_size=1 (2 bytes)
    g1[1] = (int)((td0 & 0xFFFFu) << 16);                     // tensor_dim0 lo16
    g1[2] = (int)((td0 >> 16) | ((td1 & 0xFFFFu) << 16));     // dim0 hi | dim1 lo
    g1[3] = (int)((td1 >> 16) | (tile0 << 16));               // dim1 hi | tile_dim0
    g1[4] = (int)(tile1 & 0xFFFFu);                           // tile_dim1 | tile_dim2=0
    g1[5] = (int)(unsigned)s0;                                // dim0_stride lo32
    g1[6] = (int)(unsigned)(s0 >> 32);                        // stride hi | dim1_stride lo
    g1[7] = 0;

    v4i gz = {0, 0, 0, 0};
#if defined(__clang_major__) && (__clang_major__ >= 23)
    v8i gz8 = {0, 0, 0, 0, 0, 0, 0, 0};
    __builtin_amdgcn_tensor_load_to_lds(g0, g1, gz, gz, gz8, 0);
#else
    __builtin_amdgcn_tensor_load_to_lds(g0, g1, gz, gz, 0);
#endif
}
#endif

// ---------------------------------------------------------------------------
// fp32 -> bf16 convert (vectorized)
// ---------------------------------------------------------------------------
__global__ __launch_bounds__(256) void cvt_bf16(const float* __restrict__ in,
                                                __bf16* __restrict__ out, int n) {
    int i = (blockIdx.x * blockDim.x + threadIdx.x) * 4;
    if (i + 3 < n) {
        float4 v = *(const float4*)(in + i);
        v4bf o;
        o.x = (__bf16)v.x; o.y = (__bf16)v.y; o.z = (__bf16)v.z; o.w = (__bf16)v.w;
        *(v4bf*)(out + i) = o;
    }
}

// ---------------------------------------------------------------------------
// Transpose + convert: W[k][n] (fp32, 1024x1024) -> Wt[n][k] (bf16).
// Makes every GEMM B-fragment two contiguous 16-byte runs.
// ---------------------------------------------------------------------------
__global__ __launch_bounds__(256) void transpose_cvt(const float* __restrict__ W,
                                                     __bf16* __restrict__ Wt) {
    __shared__ float tile[32][33];
    const int tx = threadIdx.x, ty = threadIdx.y;
    const int x = blockIdx.x * 32 + tx;        // n
    const int y0 = blockIdx.y * 32;            // k base
    #pragma unroll
    for (int j = ty; j < 32; j += 8)
        tile[j][tx] = W[(size_t)(y0 + j) * D_MODEL + x];
    __syncthreads();
    const int k = blockIdx.y * 32 + tx;
    const int nb = blockIdx.x * 32;
    #pragma unroll
    for (int j = ty; j < 32; j += 8)
        Wt[(size_t)(nb + j) * D_MODEL + k] = (__bf16)tile[tx][j];
}

// ---------------------------------------------------------------------------
// GEMM: C[M=4096][N=1024] = A_bf16[M][1024] @ Wt_bf16[N][1024]^T + bias
// Block: 256 threads = 8 waves in a 2(M) x 4(N) grid; wave tile 64x64
// (4x4 WMMA accumulators -> 16 WMMAs per 32-wide k-step, 8 fragment loads).
// Block tile 128(M) x 256(N). The 256x32 weight tile per k-step is DMAed
// into LDS by the TDM (double buffered); A fragments come straight from
// global (b128 per run, L2 resident).
// Epilogue modes:
//   0: bf16 out, head-split [B,H,S,Dk]             (Q, K)
//   1: bf16 out, head-split transposed [B,H,Dk,S]  (V)
//   2: fp32 out, row-major [tokens][1024]          (final projection -> d_out)
// ---------------------------------------------------------------------------
__global__ __launch_bounds__(256) void gemm_bf16(const __bf16* __restrict__ A,
                                                 const __bf16* __restrict__ Bt,
                                                 const float* __restrict__ bias,
                                                 void* __restrict__ out, int mode) {
    __shared__ __align__(16) __bf16 Bs[2][256 * 32];   // 2 x 16 KB double buffer

    const int tid   = threadIdx.x;
    const int wave  = tid >> 5;
    const int lane  = tid & 31;
    const int lhalf = lane >> 4;
    const int l16   = lane & 15;
    const int wm = wave & 1;          // 2 waves along M
    const int wn = wave >> 1;         // 4 waves along N
    const int m0   = blockIdx.x * 128 + wm * 64;
    const int nblk = blockIdx.y * 256;
    const int n0   = nblk + wn * 64;

    v8f acc[4][4] = {};

#ifdef USE_TDM
    if (wave == 0) {
        tdm_load_tile((unsigned)(uintptr_t)&Bs[0][0],
                      (unsigned long long)(uintptr_t)(Bt + (size_t)nblk * D_MODEL),
                      256);
    }
#endif

    for (int it = 0; it < D_MODEL / 32; ++it) {
        const int kk = it * 32;
        const int buf = it & 1;

#ifdef USE_TDM
        __syncthreads();   // everyone done reading buf^1 -> safe to overwrite
        if (wave == 0) {
            if (it + 1 < D_MODEL / 32) {
                tdm_load_tile((unsigned)(uintptr_t)&Bs[buf ^ 1][0],
                              (unsigned long long)(uintptr_t)
                                  (Bt + (size_t)nblk * D_MODEL + kk + 32),
                              256);
                __builtin_amdgcn_s_wait_tensorcnt(1);   // current tile landed
            } else {
                __builtin_amdgcn_s_wait_tensorcnt(0);
            }
        }
        __syncthreads();   // publish buf to all waves
#endif

        // A fragments (4 m-subtiles) from global
        v16bf a[4];
        #pragma unroll
        for (int mi = 0; mi < 4; ++mi) {
            const __bf16* p = A + (size_t)(m0 + mi * 16 + l16) * D_MODEL + kk + lhalf * 8;
            a[mi] = load_frag(p, p + 16);
            if (kk + 32 < D_MODEL) __builtin_prefetch(p + 32, 0, 1);
        }
        // B fragments (4 n-subtiles)
        v16bf b[4];
        #pragma unroll
        for (int ni = 0; ni < 4; ++ni) {
#ifdef USE_TDM
            const int nloc = wn * 64 + ni * 16 + l16;
            const __bf16* p = &Bs[buf][nloc * 32 + lhalf * 8];
#else
            const __bf16* p = Bt + (size_t)(n0 + ni * 16 + l16) * D_MODEL + kk + lhalf * 8;
#endif
            b[ni] = load_frag(p, p + 16);
        }
        #pragma unroll
        for (int mi = 0; mi < 4; ++mi)
            #pragma unroll
            for (int ni = 0; ni < 4; ++ni)
                acc[mi][ni] = __builtin_amdgcn_wmma_f32_16x16x32_bf16(
                    false, a[mi], false, b[ni], (short)0, acc[mi][ni], false, false);
    }

    // Epilogue. D-fragment: value r -> row = r + 8*lhalf, col = l16.
    #pragma unroll
    for (int mi = 0; mi < 4; ++mi) {
        #pragma unroll
        for (int ni = 0; ni < 4; ++ni) {
            const int gn = n0 + ni * 16 + l16;
            const float bb = bias[gn];
            #pragma unroll
            for (int r = 0; r < 8; ++r) {
                const int gm = m0 + mi * 16 + r + lhalf * 8;
                const float val = acc[mi][ni][r] + bb;
                if (mode == 2) {
                    ((float*)out)[(size_t)gm * D_MODEL + gn] = val;
                } else {
                    const int b  = gm >> 11;          // token / SEQ
                    const int s  = gm & (SEQ - 1);
                    const int hh = gn >> 6;           // head
                    const int d  = gn & (D_K - 1);
                    const int bh = b * N_HEADS + hh;
                    const size_t idx = (mode == 0)
                        ? ((size_t)bh * SEQ + s) * D_K + d        // [B,H,S,Dk]
                        : ((size_t)bh * D_K + d) * SEQ + s;       // [B,H,Dk,S]
                    ((__bf16*)out)[idx] = (__bf16)val;
                }
            }
        }
    }
}

// ---------------------------------------------------------------------------
// Flash attention: one wave per 16-row q tile, 8 waves (128 q rows) per block.
// Grid: (SEQ/128, B*H). Iterates keys in blocks of 32:
//   scores (16x32) = 4 WMMAs, fp32 online softmax (shfl_xor row reductions),
//   P staged bf16 through LDS to reshape D-frag -> A-frag, P@V = 4 WMMAs.
// ---------------------------------------------------------------------------
__global__ __launch_bounds__(256) void attention(const __bf16* __restrict__ Qh,
                                                 const __bf16* __restrict__ Kh,
                                                 const __bf16* __restrict__ Vt,
                                                 __bf16* __restrict__ AO) {
    __shared__ __align__(16) __bf16 Pbuf[8][16 * 32];

    const int tid   = threadIdx.x;
    const int wave  = tid >> 5;
    const int lane  = tid & 31;
    const int lhalf = lane >> 4;
    const int l16   = lane & 15;
    const int bh = blockIdx.y;
    const int q0 = blockIdx.x * 128 + wave * 16;

    const __bf16* Qb = Qh + (size_t)bh * SEQ * D_K;   // [S, Dk]
    const __bf16* Kb = Kh + (size_t)bh * SEQ * D_K;   // [S, Dk]  (B-operand for QK^T)
    const __bf16* Vb = Vt + (size_t)bh * D_K * SEQ;   // [Dk, S] (B-operand for P@V)

    // Q A-fragments for the two 32-wide d halves (loaded once, reused all loop).
    v16bf aq[2];
    #pragma unroll
    for (int dh = 0; dh < 2; ++dh) {
        const __bf16* p = Qb + (size_t)(q0 + l16) * D_K + dh * 32 + lhalf * 8;
        aq[dh] = load_frag(p, p + 16);
    }

    v8f acco[4] = {};
    float mrow[8], lrow[8];
    #pragma unroll
    for (int r = 0; r < 8; ++r) { mrow[r] = -1e30f; lrow[r] = 0.0f; }

    const float scale = 0.125f;  // 1/sqrt(64)

    for (int kb = 0; kb < SEQ; kb += 32) {
        // ---- scores: 16 q x 32 keys ----
        v8f accs[2] = {};
        #pragma unroll
        for (int nt = 0; nt < 2; ++nt) {
            const int key = kb + nt * 16 + l16;
            #pragma unroll
            for (int dh = 0; dh < 2; ++dh) {
                const __bf16* p = Kb + (size_t)key * D_K + dh * 32 + lhalf * 8;
                v16bf bk = load_frag(p, p + 16);
                accs[nt] = __builtin_amdgcn_wmma_f32_16x16x32_bf16(
                    false, aq[dh], false, bk, (short)0, accs[nt], false, false);
            }
        }

        // ---- online softmax (fp32). Row r+8*lhalf lives across the 16 lanes
        // of this half -> reduce with xor masks 1,2,4,8 (stays within half).
        float pv[2][8];
        #pragma unroll
        for (int r = 0; r < 8; ++r) {
            float s0 = accs[0][r] * scale;
            float s1 = accs[1][r] * scale;
            float mx = fmaxf(s0, s1);
            #pragma unroll
            for (int off = 1; off < 16; off <<= 1)
                mx = fmaxf(mx, __shfl_xor(mx, off, 32));
            const float mnew = fmaxf(mrow[r], mx);
            const float corr = __expf(mrow[r] - mnew);
            const float p0 = __expf(s0 - mnew);
            const float p1 = __expf(s1 - mnew);
            float rs = p0 + p1;
            #pragma unroll
            for (int off = 1; off < 16; off <<= 1)
                rs += __shfl_xor(rs, off, 32);
            lrow[r] = lrow[r] * corr + rs;
            mrow[r] = mnew;
            #pragma unroll
            for (int dt = 0; dt < 4; ++dt) acco[dt][r] *= corr;
            pv[0][r] = p0;
            pv[1][r] = p1;
        }

        // ---- stage P (16x32 bf16) in LDS to reshape D-frag -> A-frag ----
        #pragma unroll
        for (int nt = 0; nt < 2; ++nt)
            #pragma unroll
            for (int r = 0; r < 8; ++r)
                Pbuf[wave][(r + lhalf * 8) * 32 + nt * 16 + l16] = (__bf16)pv[nt][r];
        __syncthreads();

        const __bf16* pp = &Pbuf[wave][l16 * 32 + lhalf * 8];
        v16bf ap = load_frag(pp, pp + 16);

        // ---- out += P @ V for the four 16-wide d tiles ----
        #pragma unroll
        for (int dt = 0; dt < 4; ++dt) {
            const __bf16* p = Vb + (size_t)(dt * 16 + l16) * SEQ + kb + lhalf * 8;
            v16bf bv = load_frag(p, p + 16);
            acco[dt] = __builtin_amdgcn_wmma_f32_16x16x32_bf16(
                false, ap, false, bv, (short)0, acco[dt], false, false);
        }
        __syncthreads();
    }

    // ---- normalize and write attn output back to token-major bf16 ----
    const int b  = bh >> 4;
    const int hh = bh & 15;
    #pragma unroll
    for (int r = 0; r < 8; ++r) {
        const int s = q0 + r + lhalf * 8;
        const float inv = 1.0f / lrow[r];
        #pragma unroll
        for (int dt = 0; dt < 4; ++dt) {
            const float val = acco[dt][r] * inv;
            AO[((size_t)(b * SEQ + s)) * D_MODEL + hh * D_K + dt * 16 + l16] = (__bf16)val;
        }
    }
}

// ---------------------------------------------------------------------------
// Host-side orchestration
// ---------------------------------------------------------------------------
extern "C" void kernel_launch(void* const* d_in, const int* in_sizes, int n_in,
                              void* d_out, int out_size, void* d_ws, size_t ws_size,
                              hipStream_t stream) {
    (void)in_sizes; (void)n_in; (void)out_size; (void)ws_size;

    const float* X  = (const float*)d_in[0];
    const float* Wq = (const float*)d_in[1];
    const float* bq = (const float*)d_in[2];
    const float* Wk = (const float*)d_in[3];
    const float* bk = (const float*)d_in[4];
    const float* Wv = (const float*)d_in[5];
    const float* bv = (const float*)d_in[6];
    const float* Wo = (const float*)d_in[7];
    const float* bo = (const float*)d_in[8];

    char* ws = (char*)d_ws;
    const size_t MB = 1024ull * 1024ull;
    __bf16* Xb  = (__bf16*)(ws + 0 * MB);   //  8 MB  X in bf16 [tokens, 1024]
    __bf16* Wqt = (__bf16*)(ws + 8 * MB);   //  2 MB  Wq^T bf16 [out, in]
    __bf16* Wkt = (__bf16*)(ws + 10 * MB);  //  2 MB
    __bf16* Wvt = (__bf16*)(ws + 12 * MB);  //  2 MB
    __bf16* Wot = (__bf16*)(ws + 14 * MB);  //  2 MB
    __bf16* Qh  = (__bf16*)(ws + 16 * MB);  //  8 MB  [B,H,S,Dk]
    __bf16* Kh  = (__bf16*)(ws + 24 * MB);  //  8 MB  [B,H,S,Dk]
    __bf16* Vt  = (__bf16*)(ws + 32 * MB);  //  8 MB  [B,H,Dk,S]
    __bf16* AO  = (__bf16*)(ws + 40 * MB);  //  8 MB  attn out bf16 [tokens, 1024]

    // 1) convert X to bf16
    cvt_bf16<<<(TOKENS * D_MODEL) / (256 * 4), 256, 0, stream>>>(X, Xb, TOKENS * D_MODEL);

    // 2) transpose + convert the four weight matrices
    {
        dim3 tb(32, 8), tg(D_MODEL / 32, D_MODEL / 32);
        transpose_cvt<<<tg, tb, 0, stream>>>(Wq, Wqt);
        transpose_cvt<<<tg, tb, 0, stream>>>(Wk, Wkt);
        transpose_cvt<<<tg, tb, 0, stream>>>(Wv, Wvt);
        transpose_cvt<<<tg, tb, 0, stream>>>(Wo, Wot);
    }

    // 3) QKV projections straight into attention-friendly layouts
    {
        dim3 gg(TOKENS / 128, D_MODEL / 256);
        gemm_bf16<<<gg, 256, 0, stream>>>(Xb, Wqt, bq, (void*)Qh, 0);
        gemm_bf16<<<gg, 256, 0, stream>>>(Xb, Wkt, bk, (void*)Kh, 0);
        gemm_bf16<<<gg, 256, 0, stream>>>(Xb, Wvt, bv, (void*)Vt, 1);
    }

    // 4) flash attention
    attention<<<dim3(SEQ / 128, BATCH * N_HEADS), 256, 0, stream>>>(Qh, Kh, Vt, AO);

    // 5) output projection -> fp32 d_out
    {
        dim3 gg(TOKENS / 128, D_MODEL / 256);
        gemm_bf16<<<gg, 256, 0, stream>>>(AO, Wot, bo, d_out, 2);
    }
}
